// MultiHeadAttentionClassical_65481071395862
// MI455X (gfx1250) — matrix operand order repaired and checked by reading, not compiled, not run on hardware
//
#include <hip/hip_runtime.h>

#define DEV __device__ __forceinline__

typedef __attribute__((ext_vector_type(16))) __bf16 v16bf;
typedef __attribute__((ext_vector_type(8)))  __bf16 v8bf;
typedef __attribute__((ext_vector_type(4)))  __bf16 v4bf;
typedef __attribute__((ext_vector_type(8)))  float  v8f;
typedef __attribute__((ext_vector_type(4)))  float  f32x4;

// D = A(16x32 bf16) x B(32x16 bf16) + C(16x16 f32)
DEV v8f wmma_bf(v16bf a, v16bf b, v8f c) {
  return __builtin_amdgcn_wmma_f32_16x16x32_bf16(
      /*neg_a=*/false, a, /*neg_b=*/false, b,
      /*c_mod=*/(short)0, c, /*reuse_a=*/false, /*reuse_b=*/false);
}

DEV v16bf cat8(v8bf a, v8bf b) {
  return __builtin_shufflevector(a, b, 0,1,2,3,4,5,6,7,8,9,10,11,12,13,14,15);
}

DEV v8bf ld8(const __bf16* p) { return *(const v8bf*)p; }

DEV unsigned f2bf(float f) {  // round-to-nearest-even fp32 -> bf16 bits
  unsigned u = __builtin_bit_cast(unsigned, f);
  unsigned r = u + 0x7FFFu + ((u >> 16) & 1u);
  return (r >> 16) & 0xFFFFu;
}

DEV v8f vzero() { return (v8f){0.f,0.f,0.f,0.f,0.f,0.f,0.f,0.f}; }

// Problem sizes: B=4 L=2048 D=512 H=8 DK=64 ; tokens = 8192
// ------------------------------------------------------------------
// Kernel 0: one-shot fp32 -> bf16 conversion (one thread per float4).
// ------------------------------------------------------------------
__global__ __launch_bounds__(256) void k_cvt(const float* __restrict__ src,
                                             __bf16* __restrict__ dst, int n4) {
  const int i = blockIdx.x * 256 + threadIdx.x;
  if (i < n4) {
    const f32x4 v = ((const f32x4*)src)[i];
    v4bf o;
    o[0] = (__bf16)v[0]; o[1] = (__bf16)v[1];
    o[2] = (__bf16)v[2]; o[3] = (__bf16)v[3];
    ((v4bf*)dst)[i] = o;
  }
}

// ------------------------------------------------------------------
// Kernel 1: Q/K/V projections from bf16 x and bf16 weights.
//   Wave tile = 32 tokens x 64 outputs (two A tiles share each B tile).
//   Q gets the 1/sqrt(DK)=0.125 softmax scale folded in.
//   Q,K stored [B,H,L,64] bf16 ; V stored transposed [B,H,64,L] bf16.
// ------------------------------------------------------------------
__global__ __launch_bounds__(256) void k_qkv(
    const __bf16* __restrict__ xb,
    const __bf16* __restrict__ wqb, const float* __restrict__ bq,
    const __bf16* __restrict__ wkb, const float* __restrict__ bk,
    const __bf16* __restrict__ wvb, const float* __restrict__ bv,
    __bf16* __restrict__ Qo, __bf16* __restrict__ Ko, __bf16* __restrict__ Vt)
{
  const int wid  = (blockIdx.x * 256 + threadIdx.x) >> 5;
  const int lane = threadIdx.x & 31;
  const int hi   = lane >> 4;
  const int ln   = lane & 15;

  const int mat = wid / 2048;            // 0=Q 1=K 2=V
  const int rem = wid - mat * 2048;
  const int it  = rem >> 3;              // 256 token tiles of 32
  const int jt  = rem & 7;               // 8 output column tiles of 64
  const int i0  = it * 32;
  const int j0  = jt * 64;

  const __bf16* W  = (mat == 0) ? wqb : (mat == 1) ? wkb : wvb;
  const float*  bs = (mat == 0) ? bq  : (mat == 1) ? bk  : bv;

  v8f ct[2][4];
#pragma unroll
  for (int m = 0; m < 2; ++m)
#pragma unroll
    for (int s = 0; s < 4; ++s) ct[m][s] = vzero();

  const __bf16* xr0 = xb + (i0 + ln) * 512;
  const __bf16* xr1 = xb + (i0 + 16 + ln) * 512;
  for (int kc = 0; kc < 16; ++kc) {       // K loop over D=512, 32 at a time
    const int k0 = kc * 32;
    // A operands: x rows, ISA A-layout (K = e + 8*hi split segments)
    const v16bf a0 = cat8(ld8(xr0 + k0 + 8 * hi), ld8(xr0 + k0 + 16 + 8 * hi));
    const v16bf a1 = cat8(ld8(xr1 + k0 + 8 * hi), ld8(xr1 + k0 + 16 + 8 * hi));
#pragma unroll
    for (int s = 0; s < 4; ++s) {
      // B operand: column j of W^T == row j of W, split segments k0+8*hi and k0+16+8*hi
      const __bf16* wr = W + (j0 + s * 16 + ln) * 512 + k0 + 8 * hi;
      const v16bf b = cat8(ld8(wr), ld8(wr + 16));
      ct[0][s] = wmma_bf(a0, b, ct[0][s]);
      ct[1][s] = wmma_bf(a1, b, ct[1][s]);
    }
  }

  const float qscale = (mat == 0) ? 0.125f : 1.f;   // fold 1/sqrt(64) into Q
#pragma unroll
  for (int s = 0; s < 4; ++s) {
    const int j  = j0 + s * 16 + ln;      // output feature (per lane)
    const float bj = bs[j];
    const int h  = j >> 6;
    const int dk = j & 63;
#pragma unroll
    for (int m = 0; m < 2; ++m) {
      if (mat < 2) {
        __bf16* P = (mat == 0) ? Qo : Ko;
#pragma unroll
        for (int r = 0; r < 8; ++r) {
          const int tok = i0 + 16 * m + 8 * hi + r;
          const int b   = tok >> 11;
          const int l   = tok & 2047;
          P[(((b * 8 + h) * 2048 + l) << 6) + dk] =
              (__bf16)((ct[m][s][r] + bj) * qscale);
        }
      } else {
#pragma unroll
        for (int r = 0; r < 8; r += 2) {  // pack 2 consecutive tokens -> b32
          const int tok = i0 + 16 * m + 8 * hi + r;
          const int b   = tok >> 11;
          const int l   = tok & 2047;
          const unsigned lo = f2bf(ct[m][s][r]     + bj);
          const unsigned up = f2bf(ct[m][s][r + 1] + bj);
          *(unsigned*)&Vt[(((b * 8 + h) * 64 + dk) << 11) + l] = lo | (up << 16);
        }
      }
    }
  }
}

// ------------------------------------------------------------------
// Kernel 2: register-only flash attention. One wave = 16 queries of one
// (b,h), streaming 32 blocks of 64 keys. Computes S^T = K*Q^T so the
// softmax'd P is lane-aligned with the B-operand layout of
// out^T = V^T * P^T (one shfl_xor(16) per register, no LDS round trip).
// Q already carries the 1/sqrt(DK) scale. Writes O as [B,L,512] bf16.
// ------------------------------------------------------------------
__global__ __launch_bounds__(256) void k_attn(
    const __bf16* __restrict__ Qi, const __bf16* __restrict__ Ki,
    const __bf16* __restrict__ Vi, __bf16* __restrict__ O)
{
  const int wid  = (blockIdx.x * 256 + threadIdx.x) >> 5;
  const int lane = threadIdx.x & 31;
  const int hi   = lane >> 4;
  const int ln   = lane & 15;

  const int bh = wid >> 7;               // 0..31
  const int qt = wid & 127;              // 128 query tiles
  const int q0 = qt * 16;

  const __bf16* Qp = Qi + bh * (2048 * 64);
  const __bf16* Kp = Ki + bh * (2048 * 64);
  const __bf16* Vp = Vi + bh * (64 * 2048);

  // B operand (Q^T): lane = query column, K = e + 8*hi split segments (ISA layout)
  v16bf qb[2];
#pragma unroll
  for (int kk = 0; kk < 2; ++kk) {
    const __bf16* p = Qp + (q0 + ln) * 64 + kk * 32 + 8 * hi;
    qb[kk] = cat8(ld8(p), ld8(p + 16));
  }

  v8f acc[4];
#pragma unroll
  for (int dt = 0; dt < 4; ++dt) acc[dt] = vzero();
  float m_run = -__builtin_inff();
  float l_run = 0.f;

  for (int kb = 0; kb < 32; ++kb) {       // 32 blocks of 64 keys
    const int k0 = kb * 64;

    if (kb + 1 < 32) {                    // prefetch next block (L2-resident)
      __builtin_prefetch(Kp + (k0 + 64 + ln * 2) * 64, 0, 1);
      __builtin_prefetch(Vp + (ln * 4) * 2048 + k0 + 64, 0, 1);
    }

    // S^T tiles: st[t] covers keys k0+16t .. k0+16t+15 (queries along lanes)
    v8f st[4];
#pragma unroll
    for (int t = 0; t < 4; ++t) st[t] = vzero();
#pragma unroll
    for (int kk = 0; kk < 2; ++kk) {      // dk chunks of 32
#pragma unroll
      for (int t = 0; t < 4; ++t) {
        const __bf16* kr = Kp + (k0 + 16 * t + ln) * 64 + kk * 32;
        const v16bf a = cat8(ld8(kr + 8 * hi), ld8(kr + 16 + 8 * hi));
        st[t] = wmma_bf(a, qb[kk], st[t]);
      }
    }
    // block row-max (per query = per lane; one cross-half shuffle)
    float bm = -__builtin_inff();
#pragma unroll
    for (int t = 0; t < 4; ++t)
#pragma unroll
      for (int r = 0; r < 8; ++r)
        bm = fmaxf(bm, st[t][r]);
    bm = fmaxf(bm, __shfl_xor(bm, 16, 32));
    const float m_new = fmaxf(m_run, bm);
    const float alpha = __expf(m_run - m_new);
    float rs = 0.f;
#pragma unroll
    for (int t = 0; t < 4; ++t)
#pragma unroll
      for (int r = 0; r < 8; ++r) {
        st[t][r] = __expf(st[t][r] - m_new);
        rs += st[t][r];
      }
    rs += __shfl_xor(rs, 16, 32);
    l_run = l_run * alpha + rs;
    m_run = m_new;

    // P^T -> B-operand layout: swap key ranges across lane halves.
    // pb[0] = keys k0..k0+31 (st[0],st[1]); pb[1] = keys k0+32..63 (st[2],st[3])
    v16bf pb[2];
#pragma unroll
    for (int c = 0; c < 2; ++c)
#pragma unroll
      for (int r = 0; r < 8; ++r) {
        const float swl = __shfl_xor(st[2 * c][r], 16, 32);
        const float swh = __shfl_xor(st[2 * c + 1][r], 16, 32);
        pb[c][r]     = (__bf16)(hi ? swh             : st[2 * c][r]);
        pb[c][r + 8] = (__bf16)(hi ? st[2 * c + 1][r] : swl);
      }

#pragma unroll
    for (int dt = 0; dt < 4; ++dt) {      // out^T += V^T(16 x 64keys) * P^T
#pragma unroll
      for (int r = 0; r < 8; ++r) acc[dt][r] = acc[dt][r] * alpha;
      const __bf16* vr = Vp + (dt * 16 + ln) * 2048 + k0;  // V^T row = dk dim
      const v16bf va0 = cat8(ld8(vr + 16 * hi), ld8(vr + 8 + 16 * hi));
      acc[dt] = wmma_bf(va0, pb[0], acc[dt]);
      const v16bf va1 = cat8(ld8(vr + 32 + 16 * hi), ld8(vr + 40 + 16 * hi));
      acc[dt] = wmma_bf(va1, pb[1], acc[dt]);
    }
  }

  const float rl = 1.f / l_run;
  const int b   = bh >> 3;
  const int h   = bh & 7;
  const int row = b * 2048 + q0 + ln;     // token row in O[8192,512]
#pragma unroll
  for (int dt = 0; dt < 4; ++dt)
#pragma unroll
    for (int r = 0; r < 8; r += 2) {      // consecutive d -> packed b32 store
      const int d = h * 64 + dt * 16 + 8 * hi + r;
      const unsigned lo = f2bf(acc[dt][r]     * rl);
      const unsigned up = f2bf(acc[dt][r + 1] * rl);
      *(unsigned*)&O[row * 512 + d] = lo | (up << 16);
    }
}

// ------------------------------------------------------------------
// Kernel 3: output projection y = O @ wo^T + bo (fp32 out).
//   Wave tile = 32 tokens x 64 outputs, all-bf16 operands.
// ------------------------------------------------------------------
__global__ __launch_bounds__(256) void k_oproj(
    const __bf16* __restrict__ O, const __bf16* __restrict__ wob,
    const float* __restrict__ bo, float* __restrict__ Y)
{
  const int wid  = (blockIdx.x * 256 + threadIdx.x) >> 5;
  const int lane = threadIdx.x & 31;
  const int hi   = lane >> 4;
  const int ln   = lane & 15;
  const int it   = wid >> 3;              // 256 token tiles of 32
  const int jt   = wid & 7;
  const int i0   = it * 32;
  const int j0   = jt * 64;

  v8f ct[2][4];
#pragma unroll
  for (int m = 0; m < 2; ++m)
#pragma unroll
    for (int s = 0; s < 4; ++s) ct[m][s] = vzero();

  const __bf16* or0 = O + (i0 + ln) * 512;
  const __bf16* or1 = O + (i0 + 16 + ln) * 512;
  for (int kc = 0; kc < 16; ++kc) {
    const int k0 = kc * 32;
    const v16bf a0 = cat8(ld8(or0 + k0 + 8 * hi), ld8(or0 + k0 + 16 + 8 * hi));
    const v16bf a1 = cat8(ld8(or1 + k0 + 8 * hi), ld8(or1 + k0 + 16 + 8 * hi));
#pragma unroll
    for (int s = 0; s < 4; ++s) {
      const __bf16* wr = wob + (j0 + s * 16 + ln) * 512 + k0 + 8 * hi;
      const v16bf b = cat8(ld8(wr), ld8(wr + 16));
      ct[0][s] = wmma_bf(a0, b, ct[0][s]);
      ct[1][s] = wmma_bf(a1, b, ct[1][s]);
    }
  }
#pragma unroll
  for (int s = 0; s < 4; ++s) {
    const int j  = j0 + s * 16 + ln;
    const float bj = bo[j];
#pragma unroll
    for (int m = 0; m < 2; ++m)
#pragma unroll
      for (int r = 0; r < 8; ++r)
        Y[(i0 + 16 * m + 8 * hi + r) * 512 + j] = ct[m][s][r] + bj;
  }
}

extern "C" void kernel_launch(void* const* d_in, const int* in_sizes, int n_in,
                              void* d_out, int out_size, void* d_ws, size_t ws_size,
                              hipStream_t stream) {
  (void)in_sizes; (void)n_in; (void)out_size; (void)ws_size;
  const float* x  = (const float*)d_in[0];
  const float* wq = (const float*)d_in[1];
  const float* bq = (const float*)d_in[2];
  const float* wk = (const float*)d_in[3];
  const float* bk = (const float*)d_in[4];
  const float* wv = (const float*)d_in[5];
  const float* bv = (const float*)d_in[6];
  const float* wo = (const float*)d_in[7];
  const float* bo = (const float*)d_in[8];

  __bf16* Q   = (__bf16*)d_ws;           // [4,8,2048,64]  8 MB
  __bf16* K   = Q   + 4194304;           // [4,8,2048,64]  8 MB
  __bf16* Vt  = K   + 4194304;           // [4,8,64,2048]  8 MB
  __bf16* O   = Vt  + 4194304;           // [8192,512]     8 MB
  __bf16* xb  = O   + 4194304;           // [8192,512]     8 MB
  __bf16* wqb = xb  + 4194304;           // [512,512]    0.5 MB
  __bf16* wkb = wqb + 262144;
  __bf16* wvb = wkb + 262144;
  __bf16* wob = wvb + 262144;
  float*  y   = (float*)d_out;

  // One-shot fp32 -> bf16 conversions (bandwidth-bound, ~21 MB total read).
  k_cvt<<<4096, 256, 0, stream>>>(x,  xb,  1048576);  // 4M elems / 4
  k_cvt<<<256,  256, 0, stream>>>(wq, wqb, 65536);    // 256K elems / 4
  k_cvt<<<256,  256, 0, stream>>>(wk, wkb, 65536);
  k_cvt<<<256,  256, 0, stream>>>(wv, wvb, 65536);
  k_cvt<<<256,  256, 0, stream>>>(wo, wob, 65536);

  // 3*256*8 = 6144 waves -> 768 blocks of 8 waves
  k_qkv<<<768, 256, 0, stream>>>(xb, wqb, bq, wkb, bk, wvb, bv, Q, K, Vt);
  // 32*(2048/16) = 4096 waves -> 512 blocks
  k_attn<<<512, 256, 0, stream>>>(Q, K, Vt, O);
  // 256*8 = 2048 waves -> 256 blocks
  k_oproj<<<256, 256, 0, stream>>>(O, wob, bo, y);
}
